// RWKV7TimeMix_64304250356397
// MI455X (gfx1250) — matrix operand order repaired and checked by reading, not compiled, not run on hardware
//
#include <hip/hip_runtime.h>
#include <cstdint>

// ---------------------------------------------------------------------------
// RWKV7 TimeMix forward for MI455X (gfx1250, wave32, WMMA bf16->f32)
// B=4 T=1024 C=2048 H=32 N=64 ; D_DECAY=D_AAA=96 D_MV=64 D_GATE=256
// ---------------------------------------------------------------------------

typedef __attribute__((ext_vector_type(16))) __bf16 v16bf;
typedef __attribute__((ext_vector_type(8)))  float  v8f;
typedef __attribute__((ext_vector_type(4)))  int    v4i;

#define RW_B 4
#define RW_T 1024
#define RW_C 2048
#define RW_H 32
#define RW_N 64
#define RW_M (RW_B * RW_T)          // 4096 rows
#define D_DECAY 96
#define D_AAA   96
#define D_MV    64
#define D_GATE  256

#if __has_builtin(__builtin_amdgcn_global_load_async_to_lds_b128)
#define HAVE_ASYNC 1
#endif

#ifdef HAVE_ASYNC
// global -> LDS async DMA (ASYNCcnt-tracked), 16B per lane
__device__ __forceinline__ void async_b128(const __bf16* g, __bf16* l) {
    __builtin_amdgcn_global_load_async_to_lds_b128(
        (__attribute__((address_space(1))) v4i*)g,
        (__attribute__((address_space(3))) v4i*)l, 0, 0);
}
#endif

__device__ __forceinline__ void wait_async0() {
#ifdef HAVE_ASYNC
#if __has_builtin(__builtin_amdgcn_s_wait_asynccnt)
    __builtin_amdgcn_s_wait_asynccnt(0);
#else
    asm volatile("s_wait_asynccnt 0x0" ::: "memory");
#endif
#endif
}

__device__ __forceinline__ __bf16 f2bf(float f) {
    unsigned int u = __float_as_uint(f);
    unsigned int r = (u + 0x7FFFu + ((u >> 16) & 1u)) >> 16;
    unsigned short s = (unsigned short)r;
    __bf16 b;
    __builtin_memcpy(&b, &s, 2);
    return b;
}

__device__ __forceinline__ float sigmoidf_(float x) {
    return 1.0f / (1.0f + expf(-x));
}

// ---------------------------------------------------------------------------
// 1) Token-shift mix
// ---------------------------------------------------------------------------
__global__ void mix_kernel(const float* __restrict__ x,
                           const float* __restrict__ shift_in,
                           const float* __restrict__ cr, const float* __restrict__ cw,
                           const float* __restrict__ ck, const float* __restrict__ cv,
                           const float* __restrict__ ca, const float* __restrict__ cg,
                           const float* __restrict__ vfirst,
                           __bf16* __restrict__ xr, __bf16* __restrict__ xw,
                           __bf16* __restrict__ xk, __bf16* __restrict__ xv,
                           __bf16* __restrict__ xa, __bf16* __restrict__ xg,
                           float* __restrict__ shift_out,
                           float* __restrict__ vf_out) {
    size_t i = (size_t)blockIdx.x * blockDim.x + threadIdx.x;
    const size_t total = (size_t)RW_B * RW_T * RW_C;
    if (i >= total) return;
    int c = (int)(i % RW_C);
    int t = (int)((i / RW_C) % RW_T);
    int b = (int)(i / ((size_t)RW_T * RW_C));
    float xc = x[i];
    float xp = (t == 0) ? shift_in[(size_t)b * RW_C + c] : x[i - RW_C];
    float dx = xp - xc;
    xr[i] = f2bf(xc + dx * cr[c]);
    xw[i] = f2bf(xc + dx * cw[c]);
    xk[i] = f2bf(xc + dx * ck[c]);
    xv[i] = f2bf(xc + dx * cv[c]);
    xa[i] = f2bf(xc + dx * ca[c]);
    xg[i] = f2bf(xc + dx * cg[c]);
    if (t == RW_T - 1) shift_out[(size_t)b * RW_C + c] = xc;
    vf_out[i] = vfirst[i];
}

// ---------------------------------------------------------------------------
// 2) f32 -> bf16 cast (flat), optional activation. mode: 0=id 1=tanh 2=sigmoid
// ---------------------------------------------------------------------------
__global__ void act_cast_kernel(const float* __restrict__ src,
                                __bf16* __restrict__ dst, size_t n, int mode) {
    size_t i = (size_t)blockIdx.x * blockDim.x + threadIdx.x;
    if (i >= n) return;
    float v = src[i];
    if (mode == 1) v = tanhf(v);
    else if (mode == 2) v = sigmoidf_(v);
    dst[i] = f2bf(v);
}

// ---------------------------------------------------------------------------
// 3) Transpose + cast: in [R,Cc] f32 -> out [Cc,R] bf16 (small weights)
// ---------------------------------------------------------------------------
__global__ void transpose_cast_kernel(const float* __restrict__ src,
                                      __bf16* __restrict__ dst, int R, int Cc) {
    size_t i = (size_t)blockIdx.x * blockDim.x + threadIdx.x;
    if (i >= (size_t)R * Cc) return;
    int r = (int)(i / Cc);
    int c = (int)(i % Cc);
    dst[(size_t)c * R + r] = f2bf(src[i]);
}

// ---------------------------------------------------------------------------
// 4) WMMA GEMM: Out[M,N] = A[M,K] * W[N,K]^T (bf16 row-major along K, f32 out)
//    128x128 tile, K-stage 64, double-buffered LDS filled by
//    GLOBAL_LOAD_ASYNC_TO_LDS_B128 (ASYNCcnt), 1 wait + 1 barrier per stage,
//    16 v_wmma_f32_16x16x32_bf16 per wave per stage.
//    NG/KG template flags remove N/K edge guards for full tiles.
// ---------------------------------------------------------------------------
#define GT_M 128
#define GT_N 128
#define GT_K 64
#define GT_P 72   // LDS row pitch in bf16 (64 data + 8 pad): 144B, bank-clean

// Fragment gather + 8 WMMAs for one 32-wide k-step (s = 0/1) from LDS tiles.
#define GEMM_COMPUTE_STEP(ASP, BSP, S)                                          \
    {                                                                           \
        v16bf afrag[2];                                                         \
        v16bf bfrag[4];                                                         \
        _Pragma("unroll")                                                       \
        for (int i_ = 0; i_ < 2; ++i_) {                                        \
            int m_ = wm * 32 + i_ * 16 + l15;                                   \
            uint4* ap_ = (uint4*)&afrag[i_];                                    \
            ap_[0] = *(const uint4*)&(ASP)[m_ * GT_P + (S)*32 + khalf * 16];    \
            ap_[1] = *(const uint4*)&(ASP)[m_ * GT_P + (S)*32 + khalf * 16 + 8];\
        }                                                                       \
        _Pragma("unroll")                                                       \
        for (int j_ = 0; j_ < 4; ++j_) {                                        \
            int n_ = wn * 64 + j_ * 16 + l15;                                   \
            const uint4* bp_ = (const uint4*)&(BSP)[n_ * GT_P + (S)*32 + khalf * 16]; \
            uint4* fp_ = (uint4*)&bfrag[j_];                                    \
            fp_[0] = bp_[0];                                                    \
            fp_[1] = bp_[1];                                                    \
        }                                                                       \
        _Pragma("unroll")                                                       \
        for (int i_ = 0; i_ < 2; ++i_)                                          \
            _Pragma("unroll")                                                   \
            for (int j_ = 0; j_ < 4; ++j_)                                      \
                acc[i_][j_] = __builtin_amdgcn_wmma_f32_16x16x32_bf16(          \
                    false, afrag[i_], false, bfrag[j_],                         \
                    (short)0, acc[i_][j_], false, false);                       \
    }

template <bool NG, bool KG>
__global__ __launch_bounds__(256)
void gemm_bt_kernel(const __bf16* __restrict__ A, const __bf16* __restrict__ W,
                    float* __restrict__ Out, int M, int N, int K) {
    const int tid   = threadIdx.x;
    const int m0    = blockIdx.x * GT_M;
    const int n0    = blockIdx.y * GT_N;
    const int wave  = tid >> 5;
    const int lane  = tid & 31;
    const int wm    = wave & 3;     // 32-row slice
    const int wn    = wave >> 2;    // 64-col slice
    const int l15   = lane & 15;
    const int khalf = lane >> 4;
    const int srow  = tid >> 3;       // 0..31 (staging row base)
    const int skc   = (tid & 7) * 8;  // 0..56 (staging k offset, 16B chunks)
    const int KT    = (K + GT_K - 1) / GT_K;

    v8f acc[2][4] = {};

#ifdef HAVE_ASYNC
    __shared__ __bf16 As[2][GT_M * GT_P];
    __shared__ __bf16 Bs[2][GT_N * GT_P];

    auto prefetch = [&](int kt, int buf) {
        const int k0 = kt * GT_K;
        #pragma unroll
        for (int j = 0; j < 4; ++j) {
            const int row = srow + j * 32;
            const bool kin = KG ? ((k0 + skc) < K) : true;     // K%32==0 always
            const bool nin = NG ? ((n0 + row) < N) : true;
            __bf16* la = &As[buf][row * GT_P + skc];
            if (kin) {
                async_b128(A + (size_t)(m0 + row) * K + k0 + skc, la);
            } else {
                *(uint4*)la = make_uint4(0u, 0u, 0u, 0u);
            }
            __bf16* lb = &Bs[buf][row * GT_P + skc];
            if (kin && nin) {
                async_b128(W + (size_t)n0 * K + (size_t)row * K + k0 + skc, lb);
            } else {
                *(uint4*)lb = make_uint4(0u, 0u, 0u, 0u);
            }
        }
    };

    prefetch(0, 0);
    for (int kt = 0; kt < KT; ++kt) {
        const int cur = kt & 1;
        wait_async0();            // current buffer's async copies complete
        __syncthreads();          // ...and visible workgroup-wide
        if (kt + 1 < KT) prefetch(kt + 1, cur ^ 1);   // overlap with compute
        const __bf16* asp = As[cur];
        const __bf16* bsp = Bs[cur];
        GEMM_COMPUTE_STEP(asp, bsp, 0)
        GEMM_COMPUTE_STEP(asp, bsp, 1)
    }
#else
    __shared__ __bf16 As1[GT_M * GT_P];
    __shared__ __bf16 Bs1[GT_N * GT_P];

    for (int kt = 0; kt < KT; ++kt) {
        const int k0 = kt * GT_K;
        const uint4 z = make_uint4(0u, 0u, 0u, 0u);
        uint4 av[4], bv[4];
        #pragma unroll
        for (int j = 0; j < 4; ++j) {
            const int row = srow + j * 32;
            const bool kin = KG ? ((k0 + skc) < K) : true;
            const bool nin = NG ? ((n0 + row) < N) : true;
            av[j] = kin ? *(const uint4*)(A + (size_t)(m0 + row) * K + k0 + skc) : z;
            bv[j] = (kin && nin) ? *(const uint4*)(W + (size_t)(n0 + row) * K + k0 + skc) : z;
        }
        __syncthreads();
        #pragma unroll
        for (int j = 0; j < 4; ++j) {
            const int row = srow + j * 32;
            *(uint4*)&As1[row * GT_P + skc] = av[j];
            *(uint4*)&Bs1[row * GT_P + skc] = bv[j];
        }
        __syncthreads();
        GEMM_COMPUTE_STEP(As1, Bs1, 0)
        GEMM_COMPUTE_STEP(As1, Bs1, 1)
    }
#endif

    // Store: C/D layout -> row = elem + 8*khalf, col = l15
    #pragma unroll
    for (int i = 0; i < 2; ++i) {
        #pragma unroll
        for (int j = 0; j < 4; ++j) {
            int n = n0 + wn * 64 + j * 16 + l15;
            if (!NG || n < N) {
                int mbase = m0 + wm * 32 + i * 16 + khalf * 8;
                #pragma unroll
                for (int e = 0; e < 8; ++e)
                    Out[(size_t)(mbase + e) * N + n] = acc[i][j][e];
            }
        }
    }
}

// ---------------------------------------------------------------------------
// 5) kk = normalize_per_head(k * k_k). One wave per (b,t,h).
// ---------------------------------------------------------------------------
__global__ __launch_bounds__(256)
void kk_kernel(const float* __restrict__ kraw, const float* __restrict__ k_k,
               float* __restrict__ kkout, int total_heads) {
    int gw = blockIdx.x * 8 + (threadIdx.x >> 5);
    if (gw >= total_heads) return;
    int lane = threadIdx.x & 31;
    int h = gw % RW_H;
    size_t base = (size_t)gw * RW_N;
    float x0 = kraw[base + lane]      * k_k[h * RW_N + lane];
    float x1 = kraw[base + lane + 32] * k_k[h * RW_N + lane + 32];
    float ss = x0 * x0 + x1 * x1;
    #pragma unroll
    for (int o = 16; o > 0; o >>= 1) ss += __shfl_xor(ss, o, 32);
    float inv = 1.0f / fmaxf(sqrtf(ss), 1e-12f);
    kkout[base + lane]      = x0 * inv;
    kkout[base + lane + 32] = x1 * inv;
}

// ---------------------------------------------------------------------------
// 6) Fused decay / a / v-residual / k-scaling (elementwise).
// ---------------------------------------------------------------------------
__global__ void post1_kernel(const float* __restrict__ wlow,
                             const float* __restrict__ alow,
                             const float* __restrict__ vlow,
                             const float* __restrict__ kraw,
                             const float* __restrict__ vraw,
                             const float* __restrict__ vfirst,
                             const float* __restrict__ w0,
                             const float* __restrict__ a0,
                             const float* __restrict__ v0,
                             const float* __restrict__ k_a,
                             float* __restrict__ decay,
                             float* __restrict__ aout,
                             float* __restrict__ vout,
                             float* __restrict__ k2) {
    size_t i = (size_t)blockIdx.x * blockDim.x + threadIdx.x;
    const size_t total = (size_t)RW_M * RW_C;
    if (i >= total) return;
    int c = (int)(i % RW_C);
    float z  = w0[c] + wlow[i];
    float sp = log1pf(expf(-z));                  // softplus(-z)
    decay[i] = expf(-expf(-sp - 0.5f));
    float a  = sigmoidf_(a0[c] + alow[i]);
    aout[i]  = a;
    float gv = sigmoidf_(v0[c] + vlow[i]);
    float vr = vraw[i];
    vout[i]  = vr + (vfirst[i] - vr) * gv;
    k2[i]    = kraw[i] * (1.0f + (a - 1.0f) * k_a[c]);
}

// ---------------------------------------------------------------------------
// 7) WKV scan: one block per (b,h), 64x64 f32 state in LDS, T steps.
//    Pipelined: 2 barriers / 2 state passes per step. After the state update,
//    a single combined pass produces y_t = S.r_t and sa_{t+1} = -S.kk_{t+1}
//    with double-buffered per-step vectors.
// ---------------------------------------------------------------------------
__global__ __launch_bounds__(256)
void scan_kernel(const float* __restrict__ r,  const float* __restrict__ w,
                 const float* __restrict__ k,  const float* __restrict__ v,
                 const float* __restrict__ kk, const float* __restrict__ a,
                 const float* __restrict__ state_in,
                 float* __restrict__ y, float* __restrict__ state_out) {
    const int bh = blockIdx.x;
    const int b = bh / RW_H, h = bh % RW_H;
    __shared__ float st[64 * 68];          // padded pitch (bank-friendly)
    __shared__ float vecs[2][6][64];       // [buf][r,w,k,v,kk,a][n]
    __shared__ float sa_s[64];
    const int tid = threadIdx.x;

    for (int e = tid; e < 4096; e += 256) {
        int vi = e >> 6, ni = e & 63;
        st[vi * 68 + ni] = state_in[((size_t)bh * 64 + vi) * 64 + ni];
    }

    auto loadvec = [&](int t, int buf) {
        size_t base = ((size_t)(b * RW_T + t) * RW_C) + (size_t)h * RW_N;
        if (tid < 64) {
            vecs[buf][0][tid] = r[base + tid];
            vecs[buf][1][tid] = w[base + tid];
            vecs[buf][2][tid] = k[base + tid];
        } else if (tid < 128) {
            int e = tid - 64;
            vecs[buf][3][e] = v[base + e];
            vecs[buf][4][e] = kk[base + e];
            vecs[buf][5][e] = a[base + e];
        }
    };

    loadvec(0, 0);
    __syncthreads();

    const int row = tid >> 2, part = tid & 3;
    const int ro  = row * 68 + part * 16;

    {   // sa_0 = -S . kk_0
        float p = 0.0f;
        #pragma unroll
        for (int i = 0; i < 16; ++i)
            p += st[ro + i] * vecs[0][4][part * 16 + i];
        p += __shfl_xor(p, 1, 32);
        p += __shfl_xor(p, 2, 32);
        if (part == 0) sa_s[row] = -p;
    }
    __syncthreads();

    for (int t = 0; t < RW_T; ++t) {
        const int cur = t & 1, nxt = cur ^ 1;
        const bool more = (t + 1) < RW_T;

        // phase A: update state; meanwhile waves 0-3 fetch next step's vectors
        if (more) loadvec(t + 1, nxt);
        #pragma unroll
        for (int j = 0; j < 16; ++j) {
            int e = tid + j * 256;
            int vi = e >> 6, ni = e & 63;
            float s = st[vi * 68 + ni];
            st[vi * 68 + ni] = s * vecs[cur][1][ni]
                             + sa_s[vi] * (vecs[cur][4][ni] * vecs[cur][5][ni])
                             + vecs[cur][3][vi] * vecs[cur][2][ni];
        }
        __syncthreads();

        // phase B: combined y_t and sa_{t+1} in one state pass
        float q = 0.0f, p = 0.0f;
        #pragma unroll
        for (int i = 0; i < 16; ++i) {
            float sv = st[ro + i];
            q += sv * vecs[cur][0][part * 16 + i];
            if (more) p += sv * vecs[nxt][4][part * 16 + i];
        }
        q += __shfl_xor(q, 1, 32);
        q += __shfl_xor(q, 2, 32);
        p += __shfl_xor(p, 1, 32);
        p += __shfl_xor(p, 2, 32);
        if (part == 0) {
            y[((size_t)(b * RW_T + t) * RW_C) + (size_t)h * RW_N + row] = q;
            sa_s[row] = -p;
        }
        __syncthreads();
    }

    for (int e = tid; e < 4096; e += 256) {
        int vi = e >> 6, ni = e & 63;
        state_out[((size_t)bh * 64 + vi) * 64 + ni] = st[vi * 68 + ni];
    }
}

// ---------------------------------------------------------------------------
// 8) GroupNorm + r_k bonus + gate -> bf16 for Wo GEMM. One wave per (b,t,h).
// ---------------------------------------------------------------------------
__global__ __launch_bounds__(256)
void post2_kernel(const float* __restrict__ y,  const float* __restrict__ r,
                  const float* __restrict__ k2, const float* __restrict__ v,
                  const float* __restrict__ g,  const float* __restrict__ r_k,
                  const float* __restrict__ ln_w, const float* __restrict__ ln_b,
                  __bf16* __restrict__ zb, int total_heads) {
    int gw = blockIdx.x * 8 + (threadIdx.x >> 5);
    if (gw >= total_heads) return;
    int lane = threadIdx.x & 31;
    int h = gw % RW_H;
    size_t base = (size_t)gw * RW_N;
    float y0 = y[base + lane], y1 = y[base + lane + 32];
    float s = y0 + y1, s2 = y0 * y0 + y1 * y1;
    float p = r[base + lane]      * k2[base + lane]      * r_k[h * RW_N + lane]
            + r[base + lane + 32] * k2[base + lane + 32] * r_k[h * RW_N + lane + 32];
    #pragma unroll
    for (int o = 16; o > 0; o >>= 1) {
        s  += __shfl_xor(s,  o, 32);
        s2 += __shfl_xor(s2, o, 32);
        p  += __shfl_xor(p,  o, 32);
    }
    float mu  = s * (1.0f / 64.0f);
    float var = s2 * (1.0f / 64.0f) - mu * mu;
    float inv = rsqrtf(var + 64e-5f);
    float yn0 = (y0 - mu) * inv * ln_w[h * RW_N + lane]      + ln_b[h * RW_N + lane];
    float yn1 = (y1 - mu) * inv * ln_w[h * RW_N + lane + 32] + ln_b[h * RW_N + lane + 32];
    zb[base + lane]      = f2bf((yn0 + p * v[base + lane])      * g[base + lane]);
    zb[base + lane + 32] = f2bf((yn1 + p * v[base + lane + 32]) * g[base + lane + 32]);
}

// ---------------------------------------------------------------------------
// Host-side orchestration
// ---------------------------------------------------------------------------
extern "C" void kernel_launch(void* const* d_in, const int* in_sizes, int n_in,
                              void* d_out, int out_size, void* d_ws, size_t ws_size,
                              hipStream_t stream) {
    (void)in_sizes; (void)n_in; (void)out_size; (void)ws_size;

    const float* x        = (const float*)d_in[0];
    const float* shift_in = (const float*)d_in[1];
    const float* wkv_in   = (const float*)d_in[2];
    const float* v_first  = (const float*)d_in[3];
    const float* x_r      = (const float*)d_in[4];
    const float* x_w      = (const float*)d_in[5];
    const float* x_k      = (const float*)d_in[6];
    const float* x_v      = (const float*)d_in[7];
    const float* x_a      = (const float*)d_in[8];
    const float* x_g      = (const float*)d_in[9];
    const float* w0       = (const float*)d_in[10];
    const float* w1       = (const float*)d_in[11];
    const float* w2       = (const float*)d_in[12];
    const float* a0       = (const float*)d_in[13];
    const float* a1       = (const float*)d_in[14];
    const float* a2       = (const float*)d_in[15];
    const float* v0       = (const float*)d_in[16];
    const float* v1       = (const float*)d_in[17];
    const float* v2       = (const float*)d_in[18];
    const float* g1       = (const float*)d_in[19];
    const float* g2       = (const float*)d_in[20];
    const float* k_k      = (const float*)d_in[21];
    const float* k_a      = (const float*)d_in[22];
    const float* r_k      = (const float*)d_in[23];
    const float* Wr       = (const float*)d_in[24];
    const float* Wk       = (const float*)d_in[25];
    const float* Wv       = (const float*)d_in[26];
    const float* Wo       = (const float*)d_in[27];
    const float* ln_w     = (const float*)d_in[28];
    const float* ln_b     = (const float*)d_in[29];

    const size_t MC = (size_t)RW_M * RW_C;
    const size_t CC = (size_t)RW_C * RW_C;

    float* out_main  = (float*)d_out;
    float* shift_out = out_main + MC;
    float* wkv_out   = shift_out + (size_t)RW_B * RW_C;
    float* vf_out    = wkv_out + (size_t)RW_B * RW_H * RW_N * RW_N;

    char* ws = (char*)d_ws;
    size_t cur = 0;
    auto alloc = [&](size_t bytes) -> void* {
        void* p = ws + cur;
        cur += (bytes + 255) & ~(size_t)255;
        return p;
    };
    __bf16* xr_b = (__bf16*)alloc(MC * 2);
    __bf16* xw_b = (__bf16*)alloc(MC * 2);
    __bf16* xk_b = (__bf16*)alloc(MC * 2);
    __bf16* xv_b = (__bf16*)alloc(MC * 2);
    __bf16* xa_b = (__bf16*)alloc(MC * 2);
    __bf16* xg_b = (__bf16*)alloc(MC * 2);
    __bf16* Wrb  = (__bf16*)alloc(CC * 2);
    __bf16* Wkb  = (__bf16*)alloc(CC * 2);
    __bf16* Wvb  = (__bf16*)alloc(CC * 2);
    __bf16* Wob  = (__bf16*)alloc(CC * 2);
    __bf16* w1t  = (__bf16*)alloc((size_t)RW_C * D_DECAY * 2);
    __bf16* w2t  = (__bf16*)alloc((size_t)RW_C * D_DECAY * 2);
    __bf16* a1t  = (__bf16*)alloc((size_t)RW_C * D_AAA * 2);
    __bf16* a2t  = (__bf16*)alloc((size_t)RW_C * D_AAA * 2);
    __bf16* v1t  = (__bf16*)alloc((size_t)RW_C * D_MV * 2);
    __bf16* v2t  = (__bf16*)alloc((size_t)RW_C * D_MV * 2);
    __bf16* g1t  = (__bf16*)alloc((size_t)RW_C * D_GATE * 2);
    __bf16* g2t  = (__bf16*)alloc((size_t)RW_C * D_GATE * 2);
    float*  rbuf = (float*)alloc(MC * 4);
    float*  kraw = (float*)alloc(MC * 4);
    float*  vraw = (float*)alloc(MC * 4);
    float*  hw   = (float*)alloc((size_t)RW_M * D_DECAY * 4);
    float*  ha   = (float*)alloc((size_t)RW_M * D_AAA * 4);
    float*  hv   = (float*)alloc((size_t)RW_M * D_MV * 4);
    float*  hg   = (float*)alloc((size_t)RW_M * D_GATE * 4);
    __bf16* hwb  = (__bf16*)alloc((size_t)RW_M * D_DECAY * 2);
    __bf16* hab  = (__bf16*)alloc((size_t)RW_M * D_AAA * 2);
    __bf16* hvb  = (__bf16*)alloc((size_t)RW_M * D_MV * 2);
    __bf16* hgb  = (__bf16*)alloc((size_t)RW_M * D_GATE * 2);
    float*  wlow = (float*)alloc(MC * 4);
    float*  alow = (float*)alloc(MC * 4);
    float*  vlow = (float*)alloc(MC * 4);
    float*  gbuf = (float*)alloc(MC * 4);
    float*  decay= (float*)alloc(MC * 4);
    float*  abuf = (float*)alloc(MC * 4);
    float*  vbuf = (float*)alloc(MC * 4);
    float*  k2   = (float*)alloc(MC * 4);
    float*  kkb  = (float*)alloc(MC * 4);
    float*  ybuf = (float*)alloc(MC * 4);
    __bf16* zb   = (__bf16*)alloc(MC * 2);

    const int TPB = 256;
    const int mcBlocks = (int)((MC + TPB - 1) / TPB);

    mix_kernel<<<mcBlocks, TPB, 0, stream>>>(x, shift_in, x_r, x_w, x_k, x_v, x_a, x_g,
                                             v_first, xr_b, xw_b, xk_b, xv_b, xa_b, xg_b,
                                             shift_out, vf_out);

    const int ccBlocks = (int)((CC + TPB - 1) / TPB);
    act_cast_kernel<<<ccBlocks, TPB, 0, stream>>>(Wr, Wrb, CC, 0);
    act_cast_kernel<<<ccBlocks, TPB, 0, stream>>>(Wk, Wkb, CC, 0);
    act_cast_kernel<<<ccBlocks, TPB, 0, stream>>>(Wv, Wvb, CC, 0);
    act_cast_kernel<<<ccBlocks, TPB, 0, stream>>>(Wo, Wob, CC, 0);
    auto tblocks = [&](size_t n) { return (int)((n + TPB - 1) / TPB); };
    transpose_cast_kernel<<<tblocks((size_t)RW_C*D_DECAY), TPB, 0, stream>>>(w1, w1t, RW_C, D_DECAY);
    transpose_cast_kernel<<<tblocks((size_t)RW_C*D_DECAY), TPB, 0, stream>>>(w2, w2t, D_DECAY, RW_C);
    transpose_cast_kernel<<<tblocks((size_t)RW_C*D_AAA),   TPB, 0, stream>>>(a1, a1t, RW_C, D_AAA);
    transpose_cast_kernel<<<tblocks((size_t)RW_C*D_AAA),   TPB, 0, stream>>>(a2, a2t, D_AAA, RW_C);
    transpose_cast_kernel<<<tblocks((size_t)RW_C*D_MV),    TPB, 0, stream>>>(v1, v1t, RW_C, D_MV);
    transpose_cast_kernel<<<tblocks((size_t)RW_C*D_MV),    TPB, 0, stream>>>(v2, v2t, D_MV, RW_C);
    transpose_cast_kernel<<<tblocks((size_t)RW_C*D_GATE),  TPB, 0, stream>>>(g1, g1t, RW_C, D_GATE);
    transpose_cast_kernel<<<tblocks((size_t)RW_C*D_GATE),  TPB, 0, stream>>>(g2, g2t, D_GATE, RW_C);

    // GEMM dispatch: pick guard-free instantiation when tiles divide evenly.
    auto launch_gemm = [&](const __bf16* A, const __bf16* W_, float* O, int N, int K) {
        dim3 g(RW_M / GT_M, (N + GT_N - 1) / GT_N);
        const bool ng = (N % GT_N) != 0;
        const bool kg = (K % GT_K) != 0;
        if (!ng && !kg)      gemm_bt_kernel<false, false><<<g, TPB, 0, stream>>>(A, W_, O, RW_M, N, K);
        else if (ng && !kg)  gemm_bt_kernel<true,  false><<<g, TPB, 0, stream>>>(A, W_, O, RW_M, N, K);
        else if (!ng && kg)  gemm_bt_kernel<false, true ><<<g, TPB, 0, stream>>>(A, W_, O, RW_M, N, K);
        else                 gemm_bt_kernel<true,  true ><<<g, TPB, 0, stream>>>(A, W_, O, RW_M, N, K);
    };

    launch_gemm(xr_b, Wrb, rbuf, RW_C, RW_C);
    launch_gemm(xk_b, Wkb, kraw, RW_C, RW_C);
    launch_gemm(xv_b, Wvb, vraw, RW_C, RW_C);

    launch_gemm(xw_b, w1t, hw, D_DECAY, RW_C);
    act_cast_kernel<<<tblocks((size_t)RW_M*D_DECAY), TPB, 0, stream>>>(hw, hwb, (size_t)RW_M*D_DECAY, 1);
    launch_gemm(hwb, w2t, wlow, RW_C, D_DECAY);

    launch_gemm(xa_b, a1t, ha, D_AAA, RW_C);
    act_cast_kernel<<<tblocks((size_t)RW_M*D_AAA), TPB, 0, stream>>>(ha, hab, (size_t)RW_M*D_AAA, 0);
    launch_gemm(hab, a2t, alow, RW_C, D_AAA);

    launch_gemm(xv_b, v1t, hv, D_MV, RW_C);
    act_cast_kernel<<<tblocks((size_t)RW_M*D_MV), TPB, 0, stream>>>(hv, hvb, (size_t)RW_M*D_MV, 0);
    launch_gemm(hvb, v2t, vlow, RW_C, D_MV);

    launch_gemm(xg_b, g1t, hg, D_GATE, RW_C);
    act_cast_kernel<<<tblocks((size_t)RW_M*D_GATE), TPB, 0, stream>>>(hg, hgb, (size_t)RW_M*D_GATE, 2);
    launch_gemm(hgb, g2t, gbuf, RW_C, D_GATE);

    const int totalHeads = RW_M * RW_H;
    kk_kernel<<<totalHeads / 8, TPB, 0, stream>>>(kraw, k_k, kkb, totalHeads);
    post1_kernel<<<mcBlocks, TPB, 0, stream>>>(wlow, alow, vlow, kraw, vraw, v_first,
                                               w0, a0, v0, k_a, decay, abuf, vbuf, k2);

    scan_kernel<<<RW_B * RW_H, TPB, 0, stream>>>(rbuf, decay, k2, vbuf, kkb, abuf,
                                                 wkv_in, ybuf, wkv_out);

    post2_kernel<<<totalHeads / 8, TPB, 0, stream>>>(ybuf, rbuf, k2, vbuf, gbuf, r_k,
                                                     ln_w, ln_b, zb, totalHeads);
    launch_gemm(zb, Wob, out_main, RW_C, RW_C);
}
